// Transolver_17394617549107
// MI455X (gfx1250) — matrix operand order repaired and checked by reading, not compiled, not run on hardware
//
#include <hip/hip_runtime.h>

// ---------------------------------------------------------------------------
// Transolver fused pipeline for gfx1250 (MI455X), bf16 WMMA 16x16x32.
// ---------------------------------------------------------------------------

#define DIMX   256
#define INNERX 256
#define HX     8
#define CHX    32
#define MX     64
#define NBX    2
#define NPTS   65536

typedef __attribute__((ext_vector_type(16))) __bf16 v16bf;
typedef __attribute__((ext_vector_type(8)))  float  v8f;

__device__ __forceinline__ v8f wmma_bf16(v16bf a, v16bf b, v8f c) {
  return __builtin_amdgcn_wmma_f32_16x16x32_bf16(
      /*neg_a=*/false, a, /*neg_b=*/false, b,
      /*c_mod=*/(short)0, c, /*reuse_a=*/false, /*reuse_b=*/false);
}

// A fragment: matrix stored row-major [rows][stride], element = A[row][k].
// lane&15 selects row; lane>>4 selects which K-half the lane carries.
__device__ __forceinline__ v16bf frag_a(const __bf16* base, int stride,
                                        int row_base, int k_base, int lane) {
  const __bf16* p = base + (size_t)(row_base + (lane & 15)) * stride
                         + k_base + ((lane >> 4) << 3);
  v16bf a;
#pragma unroll
  for (int i = 0; i < 8; ++i) { a[i] = p[i]; a[i + 8] = p[i + 16]; }
  return a;
}

// B fragment: B^T stored row-major [cols][stride], element = B[k][col] = Bt[col][k].
__device__ __forceinline__ v16bf frag_b(const __bf16* baseT, int stride,
                                        int col_base, int k_base, int lane) {
  const __bf16* p = baseT + (size_t)(col_base + (lane & 15)) * stride
                          + k_base + ((lane >> 4) << 3);
  v16bf b;
#pragma unroll
  for (int i = 0; i < 8; ++i) { b[i] = p[i]; b[i + 8] = p[i + 16]; }
  return b;
}

// ---------------------------------------------------------------------------
// Kernel 0: weight transposition to bf16 + accumulator zeroing.
// ---------------------------------------------------------------------------
__global__ __launch_bounds__(256) void k_prep(
    const float* __restrict__ Wfx, const float* __restrict__ Wx,
    const float* __restrict__ Wsl, const float* __restrict__ Wout,
    __bf16* __restrict__ WfxT, __bf16* __restrict__ WxT,
    __bf16* __restrict__ WslT, __bf16* __restrict__ WoutT,
    float* __restrict__ tokAcc, float* __restrict__ normAcc) {
  int t = blockIdx.x * 256 + threadIdx.x;
  int stride = gridDim.x * 256;
  for (int i = t; i < DIMX * INNERX; i += stride) {
    int in = i >> 8, o = i & 255;                // W[in][o] -> Wt[o][in]
    WfxT[o * DIMX + in]  = (__bf16)Wfx[i];
    WxT[o * DIMX + in]   = (__bf16)Wx[i];
    WoutT[o * INNERX + in] = (__bf16)Wout[i];
  }
  for (int i = t; i < CHX * MX; i += stride) {
    int c = i >> 6, m = i & 63;                  // Wsl[c][m] -> WslT[m][c]
    WslT[m * CHX + c] = (__bf16)Wsl[i];
  }
  for (int i = t; i < NBX * HX * MX * CHX; i += stride) tokAcc[i] = 0.f;
  for (int i = t; i < NBX * HX * MX; i += stride) normAcc[i] = 0.f;
}

// ---------------------------------------------------------------------------
// Kernel 1: per 32-point tile: projections, soft-slice softmax, token reduce.
// ---------------------------------------------------------------------------
__global__ __launch_bounds__(256) void k_pass1(
    const float* __restrict__ x,
    const float* __restrict__ b_fx, const float* __restrict__ b_x,
    const float* __restrict__ b_slice, const float* __restrict__ temperature,
    const __bf16* __restrict__ WfxT, const __bf16* __restrict__ WxT,
    const __bf16* __restrict__ WslT,
    float* __restrict__ tokAcc, float* __restrict__ normAcc,
    __bf16* __restrict__ sw) {
  __shared__ __bf16 sA[32 * DIMX];     // 16 KB: x tile, bf16 row-major [n][k]
  __shared__ __bf16 buf2[16384];       // 32 KB: xm -> logits -> w^T (reused)
  __shared__ __bf16 buf3[INNERX * 32]; // 16 KB: fx^T  [c][n]

  const int tid  = threadIdx.x;
  const int lane = tid & 31;
  const int wv   = tid >> 5;
  const int bid  = blockIdx.x;
  const int bb   = bid / (NPTS / 32);
  const int n0   = (bid % (NPTS / 32)) * 32;

  // ---- load x tile, demote to bf16 ----
  const float4* x4 = (const float4*)(x + ((size_t)bb * NPTS + n0) * DIMX);
#pragma unroll
  for (int i = tid; i < (32 * DIMX) / 4; i += 256) {
    float4 v = x4[i];
    int e = i * 4;
    sA[e + 0] = (__bf16)v.x; sA[e + 1] = (__bf16)v.y;
    sA[e + 2] = (__bf16)v.z; sA[e + 3] = (__bf16)v.w;
  }
  __syncthreads();

  // ---- GEMM1: xm = x @ W_x  (32x256) -> buf2 row-major ----
#pragma unroll
  for (int i = 0; i < 4; ++i) {
    int t = wv * 4 + i, rt = t >> 4, ot = t & 15;
    v8f acc = {};
#pragma unroll
    for (int kb = 0; kb < 8; ++kb) {
      if (kb < 7)
        __builtin_prefetch(WxT + (size_t)(ot * 16 + (lane & 15)) * DIMX + (kb + 1) * 32, 0, 1);
      acc = wmma_bf16(frag_a(sA, DIMX, rt * 16, kb * 32, lane),
                      frag_b(WxT, DIMX, ot * 16, kb * 32, lane), acc);
    }
    int col = ot * 16 + (lane & 15);
    float bias = b_x[col];
    int rb = rt * 16 + ((lane >> 4) << 3);
#pragma unroll
    for (int r = 0; r < 8; ++r)
      buf2[(rb + r) * INNERX + col] = (__bf16)(acc[r] + bias);
  }
  __syncthreads();

  // ---- GEMM3: fx = x @ W_fx (32x256) -> buf3 transposed [c][n] ----
#pragma unroll
  for (int i = 0; i < 4; ++i) {
    int t = wv * 4 + i, rt = t >> 4, ot = t & 15;
    v8f acc = {};
#pragma unroll
    for (int kb = 0; kb < 8; ++kb)
      acc = wmma_bf16(frag_a(sA, DIMX, rt * 16, kb * 32, lane),
                      frag_b(WfxT, DIMX, ot * 16, kb * 32, lane), acc);
    int col = ot * 16 + (lane & 15);
    float bias = b_fx[col];
    int rb = rt * 16 + ((lane >> 4) << 3);
#pragma unroll
    for (int r = 0; r < 8; ++r)
      buf3[col * 32 + rb + r] = (__bf16)(acc[r] + bias);
  }

  // ---- GEMM2: logits[h] = xm_h (32x32) @ W_slice (32x64), kept in regs ----
  v8f lac[8];
#pragma unroll
  for (int i = 0; i < 8; ++i) {
    int t = wv * 8 + i, h = t >> 3, sub = t & 7, rt = sub >> 2, mt = sub & 3;
    v8f acc = {};
    acc = wmma_bf16(frag_a(buf2, INNERX, rt * 16, h * CHX, lane),
                    frag_b(WslT, CHX, mt * 16, 0, lane), acc);
    lac[i] = acc;
  }
  __syncthreads();  // all xm reads + buf3 writes done

  // write logits bf16 over buf2: [(h*32+n)*64 + m]
#pragma unroll
  for (int i = 0; i < 8; ++i) {
    int t = wv * 8 + i, h = t >> 3, sub = t & 7, rt = sub >> 2, mt = sub & 3;
    int m = mt * 16 + (lane & 15);
    float bias = b_slice[m];
    int nb_ = rt * 16 + ((lane >> 4) << 3);
#pragma unroll
    for (int r = 0; r < 8; ++r)
      buf2[(h * 32 + nb_ + r) * MX + m] = (__bf16)(lac[i][r] + bias);
  }
  __syncthreads();

  // ---- softmax over M per (h, n); one thread per row (8*32 = 256) ----
  {
    int h = tid >> 5, n = tid & 31;
    float invt = 1.0f / fminf(fmaxf(temperature[h], 0.1f), 5.0f);
    float lv[MX];
    float mx = -1e30f;
    const __bf16* lp = buf2 + (h * 32 + n) * MX;
#pragma unroll
    for (int m = 0; m < MX; ++m) { lv[m] = (float)lp[m] * invt; mx = fmaxf(mx, lv[m]); }
    float s = 0.f;
#pragma unroll
    for (int m = 0; m < MX; ++m) { lv[m] = __expf(lv[m] - mx); s += lv[m]; }
    float inv = 1.0f / s;
    __syncthreads();  // all logits reads complete before overwrite
    __bf16* swp = sw + (((size_t)(bb * HX + h)) * NPTS + n0 + n) * MX;
#pragma unroll
    for (int m = 0; m < MX; ++m) {
      float w = lv[m] * inv;
      buf2[(h * MX + m) * 32 + n] = (__bf16)w;  // w^T  [h][m][n]
      swp[m] = (__bf16)w;                       // slice_weights global [n][m]
    }
  }
  __syncthreads();

  // ---- slice_norm partials: sum over the 32 local n per (h, m) ----
  for (int idx = tid; idx < HX * MX; idx += 256) {
    int h = idx >> 6, m = idx & 63;
    float s = 0.f;
    const __bf16* p = buf2 + (h * MX + m) * 32;
#pragma unroll
    for (int n = 0; n < 32; ++n) s += (float)p[n];
    atomicAdd(normAcc + (bb * HX + h) * MX + m, s);
  }

  // ---- GEMM4: token partial = w^T (64x32) @ fx_h (32x32) -> atomic f32 ----
#pragma unroll
  for (int i = 0; i < 8; ++i) {
    int t = wv * 8 + i, h = t >> 3, sub = t & 7, gt = sub >> 1, ct = sub & 1;
    v8f acc = {};
    acc = wmma_bf16(frag_a(buf2 + h * MX * 32, 32, gt * 16, 0, lane),
                    frag_b(buf3 + h * CHX * 32, 32, ct * 16, 0, lane), acc);
    int c = ct * 16 + (lane & 15);
    int gb = gt * 16 + ((lane >> 4) << 3);
    float* dstb = tokAcc + (size_t)((bb * HX + h) * MX) * CHX;
#pragma unroll
    for (int r = 0; r < 8; ++r) atomicAdd(dstb + (gb + r) * CHX + c, acc[r]);
  }
}

// ---------------------------------------------------------------------------
// Kernel 2: normalize tokens + tiny M=64 attention per (b, h) in f32 VALU.
// ---------------------------------------------------------------------------
__global__ __launch_bounds__(256) void k_attn(
    const float* __restrict__ tokAcc, const float* __restrict__ normAcc,
    const float* __restrict__ Wq, const float* __restrict__ Wk,
    const float* __restrict__ Wv, __bf16* __restrict__ osT) {
  __shared__ float tok[MX * CHX];
  __shared__ float qs[MX * CHX], ks[MX * CHX], vs[MX * CHX];
  __shared__ float at[MX * MX];
  int bh = blockIdx.x;
  int tid = threadIdx.x;
  const float* ta = tokAcc + (size_t)bh * MX * CHX;
  const float* na = normAcc + (size_t)bh * MX;

  for (int i = tid; i < MX * CHX; i += 256) {
    int g = i >> 5;
    tok[i] = ta[i] / (na[g] + 1e-5f);
  }
  __syncthreads();
  for (int i = tid; i < MX * CHX; i += 256) {
    int g = i >> 5, c = i & 31;
    float q = 0.f, k = 0.f, v = 0.f;
#pragma unroll
    for (int j = 0; j < CHX; ++j) {
      float tv = tok[g * CHX + j];
      q += tv * Wq[j * CHX + c];
      k += tv * Wk[j * CHX + c];
      v += tv * Wv[j * CHX + c];
    }
    qs[i] = q; ks[i] = k; vs[i] = v;
  }
  __syncthreads();
  const float scale = 0.17677669529663687f;  // 32^-0.5
  for (int i = tid; i < MX * MX; i += 256) {
    int g = i >> 6, j = i & 63;
    float s = 0.f;
#pragma unroll
    for (int c = 0; c < CHX; ++c) s += qs[g * CHX + c] * ks[j * CHX + c];
    at[i] = s * scale;
  }
  __syncthreads();
  if (tid < MX) {
    float mx = -1e30f;
    for (int j = 0; j < MX; ++j) mx = fmaxf(mx, at[tid * MX + j]);
    float s = 0.f;
    for (int j = 0; j < MX; ++j) { float e = __expf(at[tid * MX + j] - mx); at[tid * MX + j] = e; s += e; }
    float inv = 1.f / s;
    for (int j = 0; j < MX; ++j) at[tid * MX + j] *= inv;
  }
  __syncthreads();
  __bf16* os = osT + (size_t)bh * CHX * MX;  // transposed [c][g]
  for (int i = tid; i < MX * CHX; i += 256) {
    int g = i >> 5, c = i & 31;
    float o = 0.f;
#pragma unroll
    for (int j = 0; j < MX; ++j) o += at[g * MX + j] * vs[j * CHX + c];
    os[c * MX + g] = (__bf16)o;
  }
}

// ---------------------------------------------------------------------------
// Kernel 3: per 64-point tile: scatter (sw @ out_slice) then @ W_out + b_out.
// ---------------------------------------------------------------------------
__global__ __launch_bounds__(256) void k_pass3(
    const __bf16* __restrict__ sw, const __bf16* __restrict__ osT,
    const __bf16* __restrict__ WoutT, const float* __restrict__ b_out,
    float* __restrict__ out) {
  __shared__ __bf16 sOx[64 * INNERX];  // 32 KB
  const int tid  = threadIdx.x;
  const int lane = tid & 31;
  const int wv   = tid >> 5;
  const int bid  = blockIdx.x;
  const int bb   = bid / (NPTS / 64);
  const int n0   = (bid % (NPTS / 64)) * 64;

  // ---- scatter GEMM: out_x_h (64xCH) = sw_h (64xM) @ out_slice_h (MxCH) ----
#pragma unroll
  for (int i = 0; i < 8; ++i) {
    int t = wv * 8 + i, h = t >> 3, sub = t & 7, rt = sub >> 1, ct = sub & 1;
    const __bf16* swbh = sw + (((size_t)(bb * HX + h)) * NPTS + n0) * MX;
    const __bf16* osbh = osT + (size_t)(bb * HX + h) * CHX * MX;
    v8f acc = {};
    acc = wmma_bf16(frag_a(swbh, MX, rt * 16, 0, lane),
                    frag_b(osbh, MX, ct * 16, 0, lane), acc);
    acc = wmma_bf16(frag_a(swbh, MX, rt * 16, 32, lane),
                    frag_b(osbh, MX, ct * 16, 32, lane), acc);
    int col = h * CHX + ct * 16 + (lane & 15);
    int rb = rt * 16 + ((lane >> 4) << 3);
#pragma unroll
    for (int r = 0; r < 8; ++r)
      sOx[(rb + r) * INNERX + col] = (__bf16)acc[r];
  }
  __syncthreads();

  // ---- output GEMM: out (64x256) = sOx (64x256) @ W_out + b_out ----
#pragma unroll
  for (int i = 0; i < 8; ++i) {
    int t = wv * 8 + i, rt = t >> 4, ot = t & 15;
    v8f acc = {};
#pragma unroll
    for (int kb = 0; kb < 8; ++kb)
      acc = wmma_bf16(frag_a(sOx, INNERX, rt * 16, kb * 32, lane),
                      frag_b(WoutT, INNERX, ot * 16, kb * 32, lane), acc);
    int col = ot * 16 + (lane & 15);
    float bias = b_out[col];
    int rb = rt * 16 + ((lane >> 4) << 3);
    float* og = out + ((size_t)bb * NPTS + n0) * DIMX;
#pragma unroll
    for (int r = 0; r < 8; ++r)
      og[(size_t)(rb + r) * DIMX + col] = acc[r] + bias;
  }
}

// ---------------------------------------------------------------------------
// Host launcher.
// ---------------------------------------------------------------------------
extern "C" void kernel_launch(void* const* d_in, const int* in_sizes, int n_in,
                              void* d_out, int out_size, void* d_ws, size_t ws_size,
                              hipStream_t stream) {
  const float* x      = (const float*)d_in[0];
  const float* W_fx   = (const float*)d_in[1];
  const float* b_fx   = (const float*)d_in[2];
  const float* W_x    = (const float*)d_in[3];
  const float* b_x    = (const float*)d_in[4];
  const float* W_sl   = (const float*)d_in[5];
  const float* b_sl   = (const float*)d_in[6];
  const float* temp   = (const float*)d_in[7];
  const float* W_q    = (const float*)d_in[8];
  const float* W_k    = (const float*)d_in[9];
  const float* W_v    = (const float*)d_in[10];
  const float* W_out  = (const float*)d_in[11];
  const float* b_out  = (const float*)d_in[12];
  float* out = (float*)d_out;

  char* ws = (char*)d_ws;
  __bf16* WfxT    = (__bf16*)(ws + 0);          // 131072 B
  __bf16* WxT     = (__bf16*)(ws + 131072);     // 131072 B
  __bf16* WoutT   = (__bf16*)(ws + 262144);     // 131072 B
  __bf16* WslT    = (__bf16*)(ws + 393216);     //   4096 B
  float*  tokAcc  = (float*) (ws + 397312);     // 131072 B
  float*  normAcc = (float*) (ws + 528384);     //   4096 B
  __bf16* osT     = (__bf16*)(ws + 532480);     //  65536 B
  __bf16* sw      = (__bf16*)(ws + 598016);     // 128 MiB

  k_prep<<<128, 256, 0, stream>>>(W_fx, W_x, W_sl, W_out,
                                  WfxT, WxT, WslT, WoutT, tokAcc, normAcc);

  k_pass1<<<NBX * (NPTS / 32), 256, 0, stream>>>(
      x, b_fx, b_x, b_sl, temp, WfxT, WxT, WslT, tokAcc, normAcc, sw);

  k_attn<<<NBX * HX, 256, 0, stream>>>(tokAcc, normAcc, W_q, W_k, W_v, osT);

  k_pass3<<<NBX * (NPTS / 64), 256, 0, stream>>>(sw, osT, WoutT, b_out, out);
}
